// RingAttentionMock_41317585387916
// MI455X (gfx1250) — matrix operand order repaired and checked by reading, not compiled, not run on hardware
//
#include <hip/hip_runtime.h>

typedef __attribute__((ext_vector_type(16))) _Float16 v16h;
typedef __attribute__((ext_vector_type(4)))  _Float16 v4h;
typedef __attribute__((ext_vector_type(8)))  float    v8f;

#define NB    2
#define NH    16
#define SEQ   4096
#define DH    64
#define QTILE 256      // query rows per block (8 waves x 32)
#define KTILE 32       // key rows per iteration
#define KPAD  72       // sK row stride (halves): 144B rows -> 16B aligned chunks
#define VPAD  40       // sV row stride (halves): 80B rows -> 16B aligned chunks

#define USE_PERMLANE 1 // v_permlane16_b32 butterflies (VALU) instead of ds_bpermute

static __device__ __forceinline__ void put2(v16h& a, int i, unsigned int u) {
  union { unsigned int u; _Float16 h[2]; } x; x.u = u;
  a[i] = x.h[0]; a[i + 1] = x.h[1];
}

static __device__ __forceinline__ v8f wmma_f16(v16h a, v16h b, v8f c) {
  return __builtin_amdgcn_wmma_f32_16x16x32_f16(false, a, false, b, (short)0, c,
                                                false, false);
}

// raw v_exp_f32 (2^x): args are always <= 0 here; deep negatives flush to 0,
// so skip libm's denormal-range fixup (cndmask/ldexp).
static __device__ __forceinline__ float fast_exp2(float x) {
#if __has_builtin(__builtin_amdgcn_exp2f)
  return __builtin_amdgcn_exp2f(x);
#else
  return exp2f(x);
#endif
}

// per-lane 16B async copy global -> LDS (ASYNCcnt-tracked, gfx1250)
static __device__ __forceinline__ void async_b128(unsigned lds_byte, const void* g) {
  asm volatile("global_load_async_to_lds_b128 %0, %1, off"
               :: "v"(lds_byte), "v"((unsigned long long)g) : "memory");
}
static __device__ __forceinline__ void wait_async0() {
  asm volatile("s_wait_asynccnt 0x0" ::: "memory");
}

// ---- xor-butterfly reductions over 16-lane halves ----
#if USE_PERMLANE
static __device__ __forceinline__ float pl16(float x, int lo, int hi) {
  int xi = __float_as_int(x);
  return __int_as_float(__builtin_amdgcn_permlane16(xi, xi, lo, hi, false, false));
}
static __device__ __forceinline__ float xmax16(float x) {
  x = fmaxf(x, pl16(x, 0x67452301, 0xEFCDAB89));   // xor 1
  x = fmaxf(x, pl16(x, 0x45670123, 0xCDEF89AB));   // xor 2
  x = fmaxf(x, pl16(x, 0x01234567, 0x89ABCDEF));   // xor 4
  x = fmaxf(x, pl16(x, 0xFEDCBA98, 0x76543210));   // xor 8
  return x;
}
static __device__ __forceinline__ float xsum16(float x) {
  x += pl16(x, 0x67452301, 0xEFCDAB89);
  x += pl16(x, 0x45670123, 0xCDEF89AB);
  x += pl16(x, 0x01234567, 0x89ABCDEF);
  x += pl16(x, 0xFEDCBA98, 0x76543210);
  return x;
}
#else
static __device__ __forceinline__ float xmax16(float x) {
  #pragma unroll
  for (int off = 8; off > 0; off >>= 1) x = fmaxf(x, __shfl_xor(x, off, 16));
  return x;
}
static __device__ __forceinline__ float xsum16(float x) {
  #pragma unroll
  for (int off = 8; off > 0; off >>= 1) x += __shfl_xor(x, off, 16);
  return x;
}
#endif

// ---------------- shared per-tile flash-attention compute ----------------
struct FAState {
  v16h qa[2][2];     // [strip][d-chunk]
  v8f  acc[2][4];    // [strip][d-tile]
  float m[2][8], l[2][8];
};

static __device__ __forceinline__ void fa_tile(FAState& st,
    const _Float16* sK, const _Float16* sV, _Float16* sPw /* 2 x 512 */,
    int kbase, int qw0, int hh, int nn)
{
  // ---- S = Q K^T : each K B-fragment feeds both row-strips (8 WMMAs) ----
  v8f s[2][2];
  s[0][0] = (v8f){}; s[0][1] = (v8f){}; s[1][0] = (v8f){}; s[1][1] = (v8f){};
  {
    const int dk = 16 * hh;
    #pragma unroll
    for (int sub = 0; sub < 2; ++sub) {        // key sub-tile (cols 16*sub+nn)
      #pragma unroll
      for (int c = 0; c < 2; ++c) {            // d chunk
        v16h kb;                               // B: lane = N(key), K(d)=16(v>>2)+8hh+2(v&3)(+1)
        #pragma unroll
        for (int v = 0; v < 8; ++v)
          put2(kb, 2*v,
               *(const unsigned int*)&sK[(16*sub + nn) * KPAD + 32*c + 16*(v>>2) + 8*hh + 2*(v&3)]);
        s[0][sub] = wmma_f16(st.qa[0][c], kb, s[0][sub]);
        s[1][sub] = wmma_f16(st.qa[1][c], kb, s[1][sub]);
      }
    }
  }

  // ---- causal mask near the diagonal (wave-uniform branch) ----
  if (kbase + KTILE - 1 > qw0) {
    #pragma unroll
    for (int sp = 0; sp < 2; ++sp) {
      #pragma unroll
      for (int r = 0; r < 8; ++r) {
        const int qrow = qw0 + 16*sp + r + 8*hh;   // C layout: row = r + 8*hh
        if (kbase + nn      > qrow) s[sp][0][r] = -1e30f;
        if (kbase + 16 + nn > qrow) s[sp][1][r] = -1e30f;
      }
    }
  }

  // ---- online softmax (log2 domain; v_exp_f32 is natively 2^x) ----
  #pragma unroll
  for (int sp = 0; sp < 2; ++sp) {
    #pragma unroll
    for (int r = 0; r < 8; ++r) {
      const float mx   = xmax16(fmaxf(s[sp][0][r], s[sp][1][r]));
      const float mnew = fmaxf(st.m[sp][r], mx);
      const float a  = fast_exp2(st.m[sp][r] - mnew);
      const float e0 = fast_exp2(s[sp][0][r] - mnew);
      const float e1 = fast_exp2(s[sp][1][r] - mnew);
      const float rs = xsum16(e0 + e1);
      st.m[sp][r] = mnew;
      st.l[sp][r] = st.l[sp][r] * a + rs;
      #pragma unroll
      for (int t = 0; t < 4; ++t) st.acc[sp][t][r] *= a;
      const int row = r + 8 * hh;
      sPw[sp * 512 + row * 32 + nn]      = (_Float16)e0;
      sPw[sp * 512 + row * 32 + 16 + nn] = (_Float16)e1;
    }
  }
  __builtin_amdgcn_wave_barrier();   // keep DS stores before DS loads

  // ---- reload P as f16 A-fragments (one per strip) ----
  v16h pa0, pa1;
  #pragma unroll
  for (int v = 0; v < 8; ++v) {
    const int klo = 16 * (v >> 2) + 8 * hh + 2 * (v & 3);
    put2(pa0, 2*v, *(const unsigned int*)&sPw[        nn * 32 + klo]);
    put2(pa1, 2*v, *(const unsigned int*)&sPw[512 +   nn * 32 + klo]);
  }

  // ---- O += P V : each V B-fragment feeds both strips (8 WMMAs) ----
  #pragma unroll
  for (int t = 0; t < 4; ++t) {
    v16h vb;  // B: lane = N(d = 16t+nn), K(key) = 16*(v>>2) + 8*hh + 2*(v&3) (+1)
    #pragma unroll
    for (int v = 0; v < 8; ++v)
      put2(vb, 2*v, *(const unsigned int*)&sV[(16*t + nn) * VPAD + 16*(v>>2) + 8*hh + 2*(v&3)]);
    st.acc[0][t] = wmma_f16(pa0, vb, st.acc[0][t]);
    st.acc[1][t] = wmma_f16(pa1, vb, st.acc[1][t]);
  }
}

static __device__ __forceinline__ void fa_init(FAState& st, const float* Qg,
                                               int qw0, int hh, int nn)
{
  const float scale = 0.125f * 1.44269504088896340736f;  // 1/sqrt(64) * log2(e)
  #pragma unroll
  for (int sp = 0; sp < 2; ++sp) {
    const float* qrow = Qg + (size_t)(qw0 + 16*sp + nn) * DH;  // A: lane = M
    #pragma unroll
    for (int v = 0; v < 8; ++v) {
      const int klo = 16 * (v >> 2) + 8 * hh + 2 * (v & 3);
      float2 f0 = *(const float2*)(qrow + klo);
      float2 f1 = *(const float2*)(qrow + 32 + klo);
      st.qa[sp][0][2*v]   = (_Float16)(f0.x * scale);
      st.qa[sp][0][2*v+1] = (_Float16)(f0.y * scale);
      st.qa[sp][1][2*v]   = (_Float16)(f1.x * scale);
      st.qa[sp][1][2*v+1] = (_Float16)(f1.y * scale);
    }
    #pragma unroll
    for (int t = 0; t < 4; ++t) st.acc[sp][t] = (v8f){};
    #pragma unroll
    for (int r = 0; r < 8; ++r) { st.m[sp][r] = -1e30f; st.l[sp][r] = 0.0f; }
  }
}

static __device__ __forceinline__ void fa_epilogue(FAState& st, float* Og,
                                                   int qw0, int hh, int nn)
{
  #pragma unroll
  for (int sp = 0; sp < 2; ++sp) {
    #pragma unroll
    for (int r = 0; r < 8; ++r) {
      const float inv = 1.0f / st.l[sp][r];
      const int row = qw0 + 16*sp + r + 8 * hh;
      #pragma unroll
      for (int t = 0; t < 4; ++t)
        Og[(size_t)row * DH + 16*t + nn] = st.acc[sp][t][r] * inv;
    }
  }
}

// ---------------- pre-pass: f32 -> f16 conversions ----------------
__global__ __launch_bounds__(256)
void cvt_k_f16(const float* __restrict__ K, _Float16* __restrict__ Kh) {
  const long long n4 = (long long)NB * NH * SEQ * DH / 4;
  for (long long i = (long long)blockIdx.x * blockDim.x + threadIdx.x; i < n4;
       i += (long long)gridDim.x * blockDim.x) {
    float4 f = ((const float4*)K)[i];
    v4h o = { (_Float16)f.x, (_Float16)f.y, (_Float16)f.z, (_Float16)f.w };
    ((v4h*)Kh)[i] = o;
  }
}

__global__ __launch_bounds__(256)
void cvt_v_t(const float* __restrict__ V, _Float16* __restrict__ Vt) {
  __shared__ _Float16 tile[64][72];
  const int sb = blockIdx.x & (SEQ / 64 - 1);
  const int bh = blockIdx.x / (SEQ / 64);
  const size_t ibase = (size_t)bh * SEQ * DH + (size_t)sb * 64 * DH;
  for (int i = threadIdx.x; i < 64 * 64; i += 256) {
    const int r = i >> 6, c = i & 63;
    tile[r][c] = (_Float16)V[ibase + (size_t)r * DH + c];
  }
  __syncthreads();
  const size_t obase = (size_t)bh * DH * SEQ + (size_t)sb * 64;
  for (int i = threadIdx.x; i < 64 * 64; i += 256) {
    const int d = i >> 6, s = i & 63;
    Vt[obase + (size_t)d * SEQ + s] = tile[s][d];
  }
}

// ---------------- main kernel: async double-buffered staging ----------------
__global__ __launch_bounds__(256, 1)
void fa_fwd_async(const float* __restrict__ Qg_, const _Float16* __restrict__ Kh_,
                  const _Float16* __restrict__ Vt_, float* __restrict__ Og_)
{
  __shared__ __align__(16) _Float16 sK[2][KTILE * KPAD];  // [key][d]
  __shared__ __align__(16) _Float16 sV[2][DH * VPAD];     // [d][key]
  __shared__ _Float16 sP[8][2][512];

  const int tid  = threadIdx.x;
  const int lane = tid & 31;
  const int wave = tid >> 5;
  const int hh   = lane >> 4;
  const int nn   = lane & 15;

  const int qt  = blockIdx.x & (SEQ / QTILE - 1);
  const int bh  = blockIdx.x / (SEQ / QTILE);
  const int q0  = qt * QTILE;
  const int qw0 = q0 + wave * 32;

  const float*    Qg  = Qg_ + (size_t)bh * SEQ * DH;
  const _Float16* KhB = Kh_ + (size_t)bh * SEQ * DH;
  const _Float16* VtB = Vt_ + (size_t)bh * DH * SEQ;
  float*          Og  = Og_ + (size_t)bh * SEQ * DH;

  FAState st;
  fa_init(st, Qg, qw0, hh, nn);

  // 256 threads copy one 16B chunk of K and one of V^T each per tile:
  // one async-b128 instruction per wave per matrix.
  auto issue = [&](int kbase, int buf) {
    {
      const int row = tid >> 3, c = (tid & 7) * 8;            // K: 32 rows x 8 chunks
      async_b128((unsigned)(size_t)&sK[buf][row * KPAD + c],
                 KhB + (size_t)(kbase + row) * DH + c);
    }
    {
      const int d = tid >> 2, c = (tid & 3) * 8;              // V^T: 64 rows x 4 chunks
      async_b128((unsigned)(size_t)&sV[buf][d * VPAD + c],
                 VtB + (size_t)d * SEQ + kbase + c);
    }
  };

  const int ntiles = (q0 + QTILE) / KTILE;   // causal truncation
  issue(0, 0);
  for (int it = 0; it < ntiles; ++it) {
    const int cur = it & 1;
    wait_async0();                 // my async writes into buf[cur] are done
    __syncthreads();               // everyone's are; prev reads of buf[cur^1] done
    if (it + 1 < ntiles) issue((it + 1) * KTILE, cur ^ 1);
    fa_tile(st, sK[cur], sV[cur], &sP[wave][0][0], it * KTILE, qw0, hh, nn);
  }

  fa_epilogue(st, Og, qw0, hh, nn);
}

// ---------------- fallback: direct f32 staging (no scratch needed) ----------------
__global__ __launch_bounds__(256, 1)
void fa_fwd_direct(const float* __restrict__ Qg_, const float* __restrict__ Kg_,
                   const float* __restrict__ Vg_, float* __restrict__ Og_)
{
  __shared__ __align__(16) _Float16 sK[KTILE * KPAD];
  __shared__ __align__(16) _Float16 sV[DH * VPAD];
  __shared__ _Float16 sP[8][2][512];

  const int tid  = threadIdx.x;
  const int lane = tid & 31;
  const int wave = tid >> 5;
  const int hh   = lane >> 4;
  const int nn   = lane & 15;

  const int qt  = blockIdx.x & (SEQ / QTILE - 1);
  const int bh  = blockIdx.x / (SEQ / QTILE);
  const int q0  = qt * QTILE;
  const int qw0 = q0 + wave * 32;

  const size_t base = (size_t)bh * SEQ * DH;
  const float* Qg = Qg_ + base;
  const float* Kg = Kg_ + base;
  const float* Vg = Vg_ + base;
  float*       Og = Og_ + base;

  FAState st;
  fa_init(st, Qg, qw0, hh, nn);

  const int ntiles = (q0 + QTILE) / KTILE;
  for (int it = 0; it < ntiles; ++it) {
    const int kbase = it * KTILE;
    __syncthreads();
    #pragma unroll
    for (int i = tid; i < KTILE * DH; i += 256) {
      const int r = i >> 6, c = i & 63;
      sK[r * KPAD + c] = (_Float16)Kg[(size_t)(kbase + r) * DH + c];
      sV[c * VPAD + r] = (_Float16)Vg[(size_t)(kbase + r) * DH + c];
    }
    __syncthreads();
    fa_tile(st, sK, sV, &sP[wave][0][0], kbase, qw0, hh, nn);
  }

  fa_epilogue(st, Og, qw0, hh, nn);
}

extern "C" void kernel_launch(void* const* d_in, const int* in_sizes, int n_in,
                              void* d_out, int out_size, void* d_ws, size_t ws_size,
                              hipStream_t stream) {
  (void)in_sizes; (void)n_in; (void)out_size;
  const float* q = (const float*)d_in[0];
  const float* k = (const float*)d_in[1];
  const float* v = (const float*)d_in[2];
  // d_in[3] is the boolean causal mask; causality is applied analytically.
  float* out = (float*)d_out;

  const int    blocks  = NB * NH * (SEQ / QTILE);          // 512
  const size_t helems  = (size_t)NB * NH * SEQ * DH;       // 8.4M per tensor
  const size_t ws_need = 2 * helems * 2;                   // Kh + Vt in f16

  if (ws_size >= ws_need) {
    _Float16* Kh = (_Float16*)d_ws;
    _Float16* Vt = Kh + helems;
    cvt_k_f16<<<2048, 256, 0, stream>>>(k, Kh);
    cvt_v_t<<<NB * NH * (SEQ / 64), 256, 0, stream>>>(v, Vt);
    fa_fwd_async<<<blocks, 256, 0, stream>>>(q, Kh, Vt, out);
  } else {
    fa_fwd_direct<<<blocks, 256, 0, stream>>>(q, k, v, out);
  }
}
